// DeepCache_44813688766883
// MI455X (gfx1250) — matrix operand order repaired and checked by reading, not compiled, not run on hardware
//
#include <hip/hip_runtime.h>
#include <hip/hip_bf16.h>

typedef __attribute__((ext_vector_type(16))) __bf16 v16bf;
typedef __attribute__((ext_vector_type(8)))  __bf16 v8bf;
typedef __attribute__((ext_vector_type(8)))  float  v8f;
typedef __attribute__((ext_vector_type(4)))  float  v4f;

#define EMB 32
// ws layout (bf16 elements): [0,32768) pc_emb, [32768,65536) addr_emb, [65536,69632) W_enc
#define WS_PC   0
#define WS_AD   32768
#define WS_W    65536
#define WS_ELEMS 69632          // * 2 bytes = 139264 bytes needed

// ---------------- one-shot table conversion: fp32 -> bf16 into d_ws ----------------
__global__ __launch_bounds__(256) void convert_tables_kernel(
    const float* __restrict__ pc_emb, const float* __restrict__ addr_emb,
    const float* __restrict__ W, __bf16* __restrict__ ws)
{
    const int i = blockIdx.x * blockDim.x + threadIdx.x;
    if (i < 32768) {
        ws[WS_PC + i] = (__bf16)pc_emb[i];
        ws[WS_AD + i] = (__bf16)addr_emb[i];
    }
    if (i < 4096)
        ws[WS_W + i] = (__bf16)W[i];
}

// ---------------- fragment loaders ----------------
// A (16x32 bf16): lanes 0-15 hold M=l, K 0..15; lanes 16-31: K 16..31 (contiguous)
__device__ __forceinline__ v16bf load_a_bf(const __bf16* __restrict__ p, int e0) {
    v8bf a0 = *(const v8bf*)(p + e0);        // 16B
    v8bf a1 = *(const v8bf*)(p + e0 + 8);    // 16B
    return __builtin_shufflevector(a0, a1, 0,1,2,3,4,5,6,7,8,9,10,11,12,13,14,15);
}
__device__ __forceinline__ v16bf load_a_f32(const float* __restrict__ p, int e0) {
    v4f f0 = *(const v4f*)(p + e0);
    v4f f1 = *(const v4f*)(p + e0 + 4);
    v4f f2 = *(const v4f*)(p + e0 + 8);
    v4f f3 = *(const v4f*)(p + e0 + 12);
    v16bf r;
#pragma unroll
    for (int i = 0; i < 4; ++i) {
        r[i] = (__bf16)f0[i]; r[4+i] = (__bf16)f1[i];
        r[8+i] = (__bf16)f2[i]; r[12+i] = (__bf16)f3[i];
    }
    return r;
}
// B (32x16 bf16): lanes 0-15 hold K=0..15, lanes 16-31 K=16..31 (contiguous), N = lane&15
__device__ __forceinline__ v16bf load_b_f32(const float* __restrict__ p, int half) {
    const float* q = p + half * 16;
    v4f f0 = *(const v4f*)(q);
    v4f f1 = *(const v4f*)(q + 4);
    v4f f2 = *(const v4f*)(q + 8);
    v4f f3 = *(const v4f*)(q + 12);
    v16bf r;
#pragma unroll
    for (int i = 0; i < 4; ++i) {
        r[i] = (__bf16)f0[i]; r[4+i] = (__bf16)f1[i];
        r[8+i] = (__bf16)f2[i]; r[12+i] = (__bf16)f3[i];
    }
    return r;
}

// ---------------- main kernel ----------------
template<bool PRE>
__global__ __launch_bounds__(256) void deepcache_wmma_kernel(
    const unsigned int* __restrict__ x32,      // int64 x as u32: [N][2 vals][2 dwords]
    const float* __restrict__ pc_f,            // [4,256,32] fp32 (fallback)
    const float* __restrict__ ad_f,            // [4,256,32] fp32 (fallback)
    const float* __restrict__ W_f,             // [32,128]   fp32 (fallback)
    const float* __restrict__ bvec,            // [32]
    const __bf16* __restrict__ tabs,           // preconverted bf16 tables (PRE path)
    float* __restrict__ out,                   // [N,64]
    int n_rows)
{
    const int lane  = threadIdx.x & 31;
    const int half  = lane >> 4;
    const int l15   = lane & 15;
    const int e0    = half ? 16 : 0;
    const int wavesPerBlock = blockDim.x >> 5;
    const int wave   = blockIdx.x * wavesPerBlock + (threadIdx.x >> 5);
    const int nWaves = gridDim.x * wavesPerBlock;
    const int nTiles = n_rows >> 4;

    const __bf16* pc_bf = tabs + WS_PC;
    const __bf16* ad_bf = tabs + WS_AD;
    const __bf16* W_bf  = tabs + WS_W;

    // Preload B fragments (W_enc^T), reused for every tile this wave processes.
    v16bf bfrag[2][4];
#pragma unroll
    for (int c = 0; c < 2; ++c) {
#pragma unroll
        for (int s = 0; s < 4; ++s) {
            if (PRE)
                bfrag[c][s] = *(const v16bf*)(W_bf + (c*16 + l15)*128 + 32*s + half*16);
            else
                bfrag[c][s] = load_b_f32(W_f + (c*16 + l15)*128 + 32*s, half);
        }
    }
    const float bias0 = bvec[l15];
    const float bias1 = bvec[16 + l15];
    const v8f zero = {};

    for (int t = wave; t < nTiles; t += nWaves) {
        const int row = t * 16 + l15;   // lanes l and l+16 both serve row M=l15
        const unsigned int pcv = __builtin_nontemporal_load(&x32[row * 4 + 0]);
        const unsigned int adv = __builtin_nontemporal_load(&x32[row * 4 + 2]);

        if (t + nWaves < nTiles)        // global_prefetch for next tile's indices
            __builtin_prefetch(&x32[(t + nWaves) * 64 + lane], 0, 0);

        v8f acc00 = zero, acc01 = zero;   // pc:   cols 0-15, 16-31
        v8f acc10 = zero, acc11 = zero;   // addr: cols 0-15, 16-31

#pragma unroll
        for (int s = 0; s < 4; ++s) {
            const unsigned pb = (pcv >> (24 - 8 * s)) & 0xFFu;  // big-endian byte s
            const unsigned ab = (adv >> (24 - 8 * s)) & 0xFFu;
            v16bf apc, aad;
            if (PRE) {
                apc = load_a_bf(pc_bf + (s * 256 + pb) * EMB, e0);
                aad = load_a_bf(ad_bf + (s * 256 + ab) * EMB, e0);
            } else {
                apc = load_a_f32(pc_f + (s * 256 + pb) * EMB, e0);
                aad = load_a_f32(ad_f + (s * 256 + ab) * EMB, e0);
            }
            acc00 = __builtin_amdgcn_wmma_f32_16x16x32_bf16(false, apc, false, bfrag[0][s],
                                                            (short)0, acc00, false, false);
            acc01 = __builtin_amdgcn_wmma_f32_16x16x32_bf16(false, apc, false, bfrag[1][s],
                                                            (short)0, acc01, false, false);
            acc10 = __builtin_amdgcn_wmma_f32_16x16x32_bf16(false, aad, false, bfrag[0][s],
                                                            (short)0, acc10, false, false);
            acc11 = __builtin_amdgcn_wmma_f32_16x16x32_bf16(false, aad, false, bfrag[1][s],
                                                            (short)0, acc11, false, false);
        }

        // Epilogue: D VGPR r holds row (r + 8*half), col = lane&15. Bias + sigmoid,
        // non-temporal stores (streamed 64MB output; keep gather tables cache-resident).
#pragma unroll
        for (int r = 0; r < 8; ++r) {
            const int orow = t * 16 + r + half * 8;
            float* o = out + orow * 64;
            float v;
            v = acc00[r] + bias0; __builtin_nontemporal_store(1.0f/(1.0f+__expf(-v)), &o[l15]);
            v = acc01[r] + bias1; __builtin_nontemporal_store(1.0f/(1.0f+__expf(-v)), &o[16+l15]);
            v = acc10[r] + bias0; __builtin_nontemporal_store(1.0f/(1.0f+__expf(-v)), &o[32+l15]);
            v = acc11[r] + bias1; __builtin_nontemporal_store(1.0f/(1.0f+__expf(-v)), &o[48+l15]);
        }
    }
}

extern "C" void kernel_launch(void* const* d_in, const int* in_sizes, int n_in,
                              void* d_out, int out_size, void* d_ws, size_t ws_size,
                              hipStream_t stream) {
    const unsigned int* x32      = (const unsigned int*)d_in[0];
    const float*        pc_emb   = (const float*)d_in[1];
    const float*        addr_emb = (const float*)d_in[2];
    const float*        W        = (const float*)d_in[3];
    const float*        bvec     = (const float*)d_in[4];
    float*              out      = (float*)d_out;

    const int n_rows = in_sizes[0] / 2;     // x flat count = N*2 int64 elements
    const int nTiles = n_rows / 16;
    const int wavesPerBlock = 8;            // 256 threads

    // ~2 tiles per wave to amortize B-fragment preload; cap by need.
    int blocks = (nTiles + wavesPerBlock * 2 - 1) / (wavesPerBlock * 2);
    if (blocks < 1) blocks = 1;

    if (ws_size >= (size_t)WS_ELEMS * sizeof(__bf16)) {
        __bf16* tabs = (__bf16*)d_ws;
        convert_tables_kernel<<<128, 256, 0, stream>>>(pc_emb, addr_emb, W, tabs);
        deepcache_wmma_kernel<true><<<blocks, 256, 0, stream>>>(
            x32, pc_emb, addr_emb, W, bvec, tabs, out, n_rows);
    } else {
        deepcache_wmma_kernel<false><<<blocks, 256, 0, stream>>>(
            x32, pc_emb, addr_emb, W, bvec, (const __bf16*)d_ws, out, n_rows);
    }
}
